// InstrumentedMHA_26061861552301
// MI455X (gfx1250) — matrix operand order repaired and checked by reading, not compiled, not run on hardware
//
#include <hip/hip_runtime.h>

// ---------------------------------------------------------------------------
// MHA layer for MI455X (gfx1250, wave32).  All matmuls on
// v_wmma_f32_16x16x32_bf16, f32 accumulation.  All f32->bf16 conversion is
// hoisted into a cold pack pass so GEMM inner loops are load+WMMA only.
// ---------------------------------------------------------------------------

#define B_    2
#define N_    2048
#define D_    1024
#define H_    16
#define HD_   64
#define NCTX_ 256

typedef __attribute__((ext_vector_type(16))) __bf16 v16bf;
typedef __attribute__((ext_vector_type(8)))  __bf16 v8bf;
typedef __attribute__((ext_vector_type(8)))  float  v8f;
typedef __attribute__((ext_vector_type(4)))  float  v4f;

union bf16x16 {
    v16bf v;
    struct { v8bf lo, hi; } p;
};

__device__ __forceinline__ v8f vzero8() {
    v8f z;
#pragma unroll
    for (int i = 0; i < 8; ++i) z[i] = 0.f;
    return z;
}

__device__ __forceinline__ v8f wmma_bf16(v16bf a, v16bf b, v8f c) {
    return __builtin_amdgcn_wmma_f32_16x16x32_bf16(
        false, a, false, b, (short)0, c, false, false);
}

// ---- A-fragment (16x32, M x K): lane&15 = row M; per-lane 8-elem K chunks
// at kb and kb+16, kb = (lane>>4)*8.  Source row-major, K contiguous. ----
__device__ __forceinline__ v16bf load_afrag(const __bf16* __restrict__ base,
                                            int stride, int lane) {
    const int row = lane & 15;
    const int kb  = (lane >> 4) << 3;
    const __bf16* p = base + (size_t)row * stride + kb;
    bf16x16 u;
    u.p.lo = *(const v8bf*)(p);
    u.p.hi = *(const v8bf*)(p + 16);
    return u.v;
}

// ---- B-fragment (32x16, K x N): lane&15 = col N; per-lane 8-elem K chunks
// at kb and kb+16, kb = (lane>>4)*8.  Source: row n holds K contiguous. ----
__device__ __forceinline__ v16bf load_bfrag(const __bf16* __restrict__ base,
                                            int stride, int lane) {
    const int col  = lane & 15;
    const int koff = (lane >> 4) << 3;
    const __bf16* p = base + (size_t)col * stride + koff;
    bf16x16 u;
    u.p.lo = *(const v8bf*)(p);
    u.p.hi = *(const v8bf*)(p + 16);
    return u.v;
}

// ---------------------------------------------------------------------------
// Pack: f32 -> bf16, 8 elements per thread (cold, memory-bound pre-pass).
// ---------------------------------------------------------------------------
__global__ __launch_bounds__(256)
void pack_bf16(const float* __restrict__ src, __bf16* __restrict__ dst, int n8) {
    const int i = blockIdx.x * blockDim.x + threadIdx.x;
    if (i >= n8) return;
    v4f a = ((const v4f*)src)[2 * i];
    v4f b = ((const v4f*)src)[2 * i + 1];
    v8bf r;
#pragma unroll
    for (int j = 0; j < 4; ++j) {
        r[j]     = (__bf16)a[j];
        r[j + 4] = (__bf16)b[j];
    }
    ((v8bf*)dst)[i] = r;
}

// ---------------------------------------------------------------------------
// Kernel 1: qkv = x @ W_in^T + b_in -> Q,K bf16 [b,h,n,hd], V gated &
// transposed bf16 [b,h,hd,n].  Wave tile 32(M) x 64(N), K step 32.
// ---------------------------------------------------------------------------
__global__ __launch_bounds__(256)
void qkv_proj_wmma(const __bf16* __restrict__ xb,        // [B*N, D] bf16
                   const __bf16* __restrict__ Winb,      // [3D, D] bf16
                   const float* __restrict__ bin,        // [3D]
                   const float* __restrict__ ctx_trust,  // [NCTX]
                   const float* __restrict__ trust_scale,// [1]
                   __bf16* __restrict__ Qb,
                   __bf16* __restrict__ Kb,
                   __bf16* __restrict__ Vt) {
    const int lane = threadIdx.x & 31;
    const int wave = threadIdx.x >> 5;
    const int m0   = blockIdx.x * 32;
    const int c0   = blockIdx.y * 512 + wave * 64;

    v8f acc[2][4];
#pragma unroll
    for (int rt = 0; rt < 2; ++rt)
#pragma unroll
        for (int t = 0; t < 4; ++t) acc[rt][t] = vzero8();

    for (int k = 0; k < D_; k += 32) {
        v16bf a0 = load_afrag(xb + (size_t)m0 * D_ + k, D_, lane);
        v16bf a1 = load_afrag(xb + (size_t)(m0 + 16) * D_ + k, D_, lane);
#pragma unroll
        for (int t = 0; t < 4; ++t) {
            v16bf b = load_bfrag(Winb + (size_t)(c0 + 16 * t) * D_ + k, D_, lane);
            acc[0][t] = wmma_bf16(a0, b, acc[0][t]);
            acc[1][t] = wmma_bf16(a1, b, acc[1][t]);
        }
    }

    const float ts = trust_scale[0];
    const int hi = lane >> 4;
    const int nc = lane & 15;
#pragma unroll
    for (int rt = 0; rt < 2; ++rt)
#pragma unroll
    for (int t = 0; t < 4; ++t) {
        const int c = c0 + 16 * t + nc;
        const float bias = bin[c];
#pragma unroll
        for (int i = 0; i < 8; ++i) {
            const int r = m0 + rt * 16 + i + 8 * hi;
            const int b = r >> 11;
            const int n = r & (N_ - 1);
            const float val = acc[rt][t][i] + bias;
            if (c < D_) {
                const int h = c >> 6, hd = c & 63;
                Qb[(((size_t)b * H_ + h) * N_ + n) * HD_ + hd] = (__bf16)val;
            } else if (c < 2 * D_) {
                const int cc = c - D_;
                const int h = cc >> 6, hd = cc & 63;
                Kb[(((size_t)b * H_ + h) * N_ + n) * HD_ + hd] = (__bf16)val;
            } else {
                const int cc = c - 2 * D_;
                const int h = cc >> 6, hd = cc & 63;
                float g = 1.f;
                if (n < NCTX_) g = 1.f / (1.f + __expf(-ts * ctx_trust[n]));
                Vt[(((size_t)b * H_ + h) * HD_ + hd) * N_ + n] = (__bf16)(val * g);
            }
        }
    }
}

// ---------------------------------------------------------------------------
// Kernel 2: flash attention.  One wave per 32-query tile; 32-key steps.
// K/V fragments amortized over 32 queries: 16 WMMA per step.
// ---------------------------------------------------------------------------
__global__ __launch_bounds__(128)
void attn_wmma(const __bf16* __restrict__ Qb,
               const __bf16* __restrict__ Kb,
               const __bf16* __restrict__ Vt,
               const float* __restrict__ ctx_ppr,
               const float* __restrict__ log_ppr_alpha,
               const unsigned char* __restrict__ mask,   // [N,N] bool
               __bf16* __restrict__ Ob) {                // [B*N, D]
    const int lane = threadIdx.x & 31;
    const int wave = threadIdx.x >> 5;
    const int tile = blockIdx.x * 4 + wave;              // [0, B*H*N/32)
    const int q0   = (tile & (N_ / 32 - 1)) * 32;
    const int bh   = tile >> 6;                          // b*H + h
    const int bidx = bh >> 4;
    const int h    = bh & 15;

    __shared__ __bf16 pbuf[4][32 * 32];
    __bf16* pl = pbuf[wave];

    v16bf q[2][2];
#pragma unroll
    for (int rt = 0; rt < 2; ++rt) {
        const __bf16* Qbase = Qb + ((size_t)bh * N_ + q0 + rt * 16) * HD_;
        q[rt][0] = load_afrag(Qbase, HD_, lane);
        q[rt][1] = load_afrag(Qbase + 32, HD_, lane);
    }

    const float alpha = log_ppr_alpha[0];
    const float scale = 0.125f;                          // HD^-0.5
    const int hi = lane >> 4;
    const int nc = lane & 15;
    const int kb = hi << 3;

    float m[2][8], l[2][8];
    v8f o[2][4];
#pragma unroll
    for (int rt = 0; rt < 2; ++rt) {
#pragma unroll
        for (int i = 0; i < 8; ++i) { m[rt][i] = -__builtin_inff(); l[rt][i] = 0.f; }
#pragma unroll
        for (int t = 0; t < 4; ++t) o[rt][t] = vzero8();
    }

    for (int kk = 0; kk < N_; kk += 32) {
        const __bf16* Kbase = Kb + ((size_t)bh * N_ + kk) * HD_;
        if (kk + 32 < N_) {
            __builtin_prefetch((const void*)(Kbase + 32 * HD_), 0, 1);
            __builtin_prefetch((const void*)(Vt + (size_t)bh * HD_ * N_ + kk + 32), 0, 1);
        }
        // ---- S = Q K^T ----
        v16bf kf[2][2];
#pragma unroll
        for (int kt = 0; kt < 2; ++kt) {
            kf[kt][0] = load_bfrag(Kbase + (size_t)kt * 16 * HD_, HD_, lane);
            kf[kt][1] = load_bfrag(Kbase + (size_t)kt * 16 * HD_ + 32, HD_, lane);
        }
        v8f s[2][2];
#pragma unroll
        for (int rt = 0; rt < 2; ++rt)
#pragma unroll
            for (int kt = 0; kt < 2; ++kt) {
                v8f acc = vzero8();
                acc = wmma_bf16(q[rt][0], kf[kt][0], acc);
                acc = wmma_bf16(q[rt][1], kf[kt][1], acc);
                s[rt][kt] = acc;
            }

        // ---- scale + ppr column bias (per-lane column, shared by rows) ----
        const int c0k = kk + nc;
        const int c1k = kk + 16 + nc;
        const float bias0 = (c0k < NCTX_) ? alpha * __logf(fmaxf(ctx_ppr[c0k], 1e-8f)) : 0.f;
        const float bias1 = (c1k < NCTX_) ? alpha * __logf(fmaxf(ctx_ppr[c1k], 1e-8f)) : 0.f;

#pragma unroll
        for (int rt = 0; rt < 2; ++rt) {
            float rmax[8];
#pragma unroll
            for (int i = 0; i < 8; ++i) {
                const int r = q0 + rt * 16 + i + 8 * hi;
                float v0 = s[rt][0][i] * scale + bias0;
                float v1 = s[rt][1][i] * scale + bias1;
                if (mask[(size_t)r * N_ + c0k]) v0 = -__builtin_inff();
                if (mask[(size_t)r * N_ + c1k]) v1 = -__builtin_inff();
                s[rt][0][i] = v0; s[rt][1][i] = v1;
                rmax[i] = fmaxf(v0, v1);
            }
#pragma unroll
            for (int d = 1; d < 16; d <<= 1)
#pragma unroll
                for (int i = 0; i < 8; ++i)
                    rmax[i] = fmaxf(rmax[i], __shfl_xor(rmax[i], d, 32));

            float sf[8], rsum[8];
#pragma unroll
            for (int i = 0; i < 8; ++i) {
                const float mn = fmaxf(m[rt][i], rmax[i]);
                sf[i] = __expf(m[rt][i] - mn);
                m[rt][i] = mn;
                const float p0 = __expf(s[rt][0][i] - mn);
                const float p1 = __expf(s[rt][1][i] - mn);
                s[rt][0][i] = p0; s[rt][1][i] = p1;
                rsum[i] = p0 + p1;
            }
#pragma unroll
            for (int d = 1; d < 16; d <<= 1)
#pragma unroll
                for (int i = 0; i < 8; ++i)
                    rsum[i] += __shfl_xor(rsum[i], d, 32);
#pragma unroll
            for (int i = 0; i < 8; ++i) l[rt][i] = l[rt][i] * sf[i] + rsum[i];
#pragma unroll
            for (int t = 0; t < 4; ++t)
#pragma unroll
                for (int i = 0; i < 8; ++i) o[rt][t][i] *= sf[i];
        }

        // ---- transpose P through LDS: C/D layout -> A-fragment layout ----
        __syncthreads();
#pragma unroll
        for (int rt = 0; rt < 2; ++rt)
#pragma unroll
            for (int i = 0; i < 8; ++i) {
                const int r = rt * 16 + i + 8 * hi;
                pl[r * 32 + nc]      = (__bf16)s[rt][0][i];
                pl[r * 32 + 16 + nc] = (__bf16)s[rt][1][i];
            }
        __syncthreads();
        v16bf pa[2];
#pragma unroll
        for (int rt = 0; rt < 2; ++rt) {
            bf16x16 u;
            u.p.lo = *(const v8bf*)(pl + (rt * 16 + nc) * 32 + kb);
            u.p.hi = *(const v8bf*)(pl + (rt * 16 + nc) * 32 + kb + 16);
            pa[rt] = u.v;
        }

        // ---- O += P V (V^T tiles contiguous per lane; reused across rt) ----
#pragma unroll
        for (int t = 0; t < 4; ++t) {
            v16bf vb = load_bfrag(Vt + ((size_t)bh * HD_ + t * 16) * N_ + kk, N_, lane);
            o[0][t] = wmma_bf16(pa[0], vb, o[0][t]);
            o[1][t] = wmma_bf16(pa[1], vb, o[1][t]);
        }
    }

    // ---- normalize and store (concat-heads layout [b*N+n, h*64+hd]) ----
#pragma unroll
    for (int rt = 0; rt < 2; ++rt)
#pragma unroll
        for (int t = 0; t < 4; ++t)
#pragma unroll
            for (int i = 0; i < 8; ++i) {
                const int r  = q0 + rt * 16 + i + 8 * hi;
                const int hd = t * 16 + nc;
                Ob[((size_t)bidx * N_ + r) * D_ + h * HD_ + hd] =
                    (__bf16)(o[rt][t][i] / l[rt][i]);
            }
}

// ---------------------------------------------------------------------------
// Kernel 3: out = O @ W_out^T + b_out   (fp32 output).  Wave tile 32x64.
// ---------------------------------------------------------------------------
__global__ __launch_bounds__(256)
void out_proj_wmma(const __bf16* __restrict__ Ob,    // [4096, 1024] bf16
                   const __bf16* __restrict__ Woutb, // [1024, 1024] bf16
                   const float* __restrict__ bout,
                   float* __restrict__ out) {
    const int lane = threadIdx.x & 31;
    const int wave = threadIdx.x >> 5;
    const int m0   = blockIdx.x * 32;
    const int c0   = blockIdx.y * 512 + wave * 64;

    v8f acc[2][4];
#pragma unroll
    for (int rt = 0; rt < 2; ++rt)
#pragma unroll
        for (int t = 0; t < 4; ++t) acc[rt][t] = vzero8();

    for (int k = 0; k < D_; k += 32) {
        v16bf a0 = load_afrag(Ob + (size_t)m0 * D_ + k, D_, lane);
        v16bf a1 = load_afrag(Ob + (size_t)(m0 + 16) * D_ + k, D_, lane);
#pragma unroll
        for (int t = 0; t < 4; ++t) {
            v16bf b = load_bfrag(Woutb + (size_t)(c0 + 16 * t) * D_ + k, D_, lane);
            acc[0][t] = wmma_bf16(a0, b, acc[0][t]);
            acc[1][t] = wmma_bf16(a1, b, acc[1][t]);
        }
    }

    const int hi = lane >> 4;
    const int nc = lane & 15;
#pragma unroll
    for (int rt = 0; rt < 2; ++rt)
#pragma unroll
        for (int t = 0; t < 4; ++t) {
            const int c = c0 + 16 * t + nc;
            const float bb = bout[c];
#pragma unroll
            for (int i = 0; i < 8; ++i)
                out[(size_t)(m0 + rt * 16 + i + 8 * hi) * D_ + c] = acc[rt][t][i] + bb;
        }
}

// ---------------------------------------------------------------------------
extern "C" void kernel_launch(void* const* d_in, const int* in_sizes, int n_in,
                              void* d_out, int out_size, void* d_ws, size_t ws_size,
                              hipStream_t stream) {
    (void)in_sizes; (void)n_in; (void)out_size; (void)ws_size;
    const float* x          = (const float*)d_in[0];
    const float* ctx_ppr    = (const float*)d_in[1];
    const float* ctx_trust  = (const float*)d_in[2];
    /* d_in[3] = n_ctx (compile-time constant 256) */
    const unsigned char* am = (const unsigned char*)d_in[4];
    const float* Win        = (const float*)d_in[5];
    const float* bin        = (const float*)d_in[6];
    const float* Wout       = (const float*)d_in[7];
    const float* bout       = (const float*)d_in[8];
    const float* lpa        = (const float*)d_in[9];
    const float* ts         = (const float*)d_in[10];

    // workspace layout (bf16 elements)
    const size_t n_x    = (size_t)B_ * N_ * D_;      // 4M
    const size_t n_win  = (size_t)3 * D_ * D_;       // 3M
    const size_t n_wout = (size_t)D_ * D_;           // 1M
    const size_t n_qkv  = (size_t)B_ * H_ * N_ * HD_;// 4M each
    __bf16* xb    = (__bf16*)d_ws;
    __bf16* Winb  = xb + n_x;
    __bf16* Woutb = Winb + n_win;
    __bf16* Qb    = Woutb + n_wout;
    __bf16* Kb    = Qb + n_qkv;
    __bf16* Vt    = Kb + n_qkv;
    __bf16* Ob    = Vt + n_qkv;

    // cold pack pass: ~34 MB read -> ~1.5 us at 23.3 TB/s
    pack_bf16<<<dim3((int)(n_x / 8 / 256)), 256, 0, stream>>>(x, xb, (int)(n_x / 8));
    pack_bf16<<<dim3((int)(n_win / 8 / 256)), 256, 0, stream>>>(Win, Winb, (int)(n_win / 8));
    pack_bf16<<<dim3((int)(n_wout / 8 / 256)), 256, 0, stream>>>(Wout, Woutb, (int)(n_wout / 8));

    // 4096/32 = 128 M-tiles; 3072 cols = 6 groups of (8 waves * 64)
    qkv_proj_wmma<<<dim3(128, 6), 256, 0, stream>>>(xb, Winb, bin, ctx_trust, ts,
                                                    Qb, Kb, Vt);
    // B*H*(N/32) = 2048 tiles, 4 waves per block
    attn_wmma<<<dim3(512), 128, 0, stream>>>(Qb, Kb, Vt, ctx_ppr, lpa, am, Ob);
    // 1024 cols = 2 groups of (8 waves * 64)
    out_proj_wmma<<<dim3(128, 2), 256, 0, stream>>>(Ob, Woutb, bout, (float*)d_out);
}
